// SelfAlign_11458972746266
// MI455X (gfx1250) — compile-verified, matrix operand order checked by reading, not run on hardware
//
#include <hip/hip_runtime.h>

// ---------------------------------------------------------------------------
// SelfAlign block for MI455X (gfx1250, wave32, WMMA f32_16x16x32_f16).
//
// C=1024 context, BATCH=16, D=128. All heavy einsums run on v_wmma with f16
// operands / f32 accumulation. Every WMMA B-operand is staged N-major
// (K contiguous per row) so both A and B fragments are two 16-byte loads/lane.
// The K=1024 GEMMs use 2x2 register tiling per wave (32x32 output / wave):
// 4 fragment loads feed 4 WMMAs, halving L2 operand traffic per matrix op.
// ---------------------------------------------------------------------------

#define CC    1024
#define BATCH 16
#define DD    128
#define NEGF  (-1e30f)

typedef __attribute__((ext_vector_type(16))) _Float16 v16h;
typedef __attribute__((ext_vector_type(8)))  _Float16 v8h;
typedef __attribute__((ext_vector_type(8)))  float    v8f;

// ---------------------------------------------------------------------------
// WMMA helpers (ISA 7.12.2 layouts)
// ---------------------------------------------------------------------------
__device__ __forceinline__ v8f wmma(v16h a, v16h b, v8f c) {
  // (neg_a, A, neg_b, B, c_mod, C, reuse_a, reuse_b)
  return __builtin_amdgcn_wmma_f32_16x16x32_f16(false, a, false, b, (short)0, c,
                                                false, false);
}

// Load a 16x32 f16 fragment from row-major memory (ld in f16 elements).
// A-operand: lane = M row. B-operand: same code against an N-major buffer
// (lane = N column of the mathematical B), per the mirrored ISA layout.
// lanes 0-15: elems 0-7 = K[0..7],  8-15 = K[16..23]
// lanes 16-31: elems 0-7 = K[8..15], 8-15 = K[24..31]
__device__ __forceinline__ v16h load_frag16(const _Float16* __restrict__ base, int ld) {
  const int lane = threadIdx.x & 31;
  const int row  = lane & 15;
  const int kb   = (lane >> 4) << 3;
  const v8h lo = *reinterpret_cast<const v8h*>(base + (size_t)row * ld + kb);
  const v8h hi = *reinterpret_cast<const v8h*>(base + (size_t)row * ld + kb + 16);
  v16h f;
#pragma unroll
  for (int i = 0; i < 8; ++i) { f[i] = lo[i]; f[8 + i] = hi[i]; }
  return f;
}

// D tile (16x16 f32): lane = N + 16*hi ; VGPR v -> row (v + 8*hi)
#define TILE_EPILOGUE(lane, n, mh)            \
  const int lane = threadIdx.x & 31;          \
  const int n    = lane & 15;                 \
  const int mh   = (lane >> 4) << 3;

// ---------------------------------------------------------------------------
// K0a: f32 -> f16 elementwise convert
// ---------------------------------------------------------------------------
__global__ void k_cvt(const float* __restrict__ x, _Float16* __restrict__ y, int n) {
  int i = blockIdx.x * 256 + threadIdx.x;
  if (i < n) y[i] = (_Float16)x[i];
}

// K0b: hT[b][d][c] = (f16) h[c][b][d]   (N-major B-operand for h_tmp GEMM)
__global__ void k_h_transpose(const float* __restrict__ h, _Float16* __restrict__ hT) {
  int idx = blockIdx.x * 256 + threadIdx.x;        // 16*128*1024 = 2M
  int c = idx & (CC - 1);
  int d = (idx >> 10) & (DD - 1);
  int b = idx >> 17;
  hT[idx] = (_Float16)h[((size_t)c * BATCH + b) * DD + d];
}

// ---------------------------------------------------------------------------
// K1: h1/h2 = relu(h @ W^T + bias), written (b, c, e) f16.
// grid (C/16, BATCH), block 256 = 8 waves over e-tiles.
// ---------------------------------------------------------------------------
__global__ void k_h12(const _Float16* __restrict__ hh,   // (c,b,d) f16
                      const _Float16* __restrict__ W1h,  // (e,d)  N-major
                      const _Float16* __restrict__ W2h,
                      const float* __restrict__ b1, const float* __restrict__ b2,
                      _Float16* __restrict__ h1h,        // (b,c,e)
                      _Float16* __restrict__ h2h) {
  const int wave = threadIdx.x >> 5;
  const int c0 = blockIdx.x * 16;
  const int b  = blockIdx.y;
  const int e0 = wave * 16;
  const _Float16* Abase = hh + ((size_t)c0 * BATCH + b) * DD;  // row stride 16*128
  v8f acc1 = {}, acc2 = {};
#pragma unroll
  for (int k = 0; k < DD; k += 32) {
    v16h a  = load_frag16(Abase + k, BATCH * DD);
    acc1 = wmma(a, load_frag16(W1h + (size_t)e0 * DD + k, DD), acc1);
    acc2 = wmma(a, load_frag16(W2h + (size_t)e0 * DD + k, DD), acc2);
  }
  TILE_EPILOGUE(lane, n, mh)
  const float bb1 = b1[e0 + n], bb2 = b2[e0 + n];
  _Float16* o1 = h1h + ((size_t)b * CC + c0) * DD + e0;
  _Float16* o2 = h2h + ((size_t)b * CC + c0) * DD + e0;
#pragma unroll
  for (int v = 0; v < 8; ++v) {
    o1[(size_t)(mh + v) * DD + n] = (_Float16)fmaxf(acc1[v] + bb1, 0.0f);
    o2[(size_t)(mh + v) * DD + n] = (_Float16)fmaxf(acc2[v] + bb2, 0.0f);
  }
}

// ---------------------------------------------------------------------------
// K2: row softmax over j with pad mask, f32 (b,i,j) -> f16 (b,i,j).
// One wave per row; row cached in 32 VGPRs; wave32 shfl_xor reductions.
// grid (C/8, BATCH), block 256.
// ---------------------------------------------------------------------------
__global__ void k_row_softmax(const float* __restrict__ X, const int* __restrict__ mask,
                              _Float16* __restrict__ Y) {
  const int wave = threadIdx.x >> 5, lane = threadIdx.x & 31;
  const int b = blockIdx.y;
  const int i = blockIdx.x * 8 + wave;
  const float* row  = X + ((size_t)b * CC + i) * CC;
  const int*   mrow = mask + (size_t)b * CC;
  float v[32];
  float m = NEGF;
#pragma unroll
  for (int t = 0; t < 32; ++t) {
    int j = lane + t * 32;
    float x = (mrow[j] != 0) ? row[j] : NEGF;
    v[t] = x;
    m = fmaxf(m, x);
  }
#pragma unroll
  for (int s = 16; s > 0; s >>= 1) m = fmaxf(m, __shfl_xor(m, s, 32));
  float sum = 0.0f;
#pragma unroll
  for (int t = 0; t < 32; ++t) { v[t] = __expf(v[t] - m); sum += v[t]; }
#pragma unroll
  for (int s = 16; s > 0; s >>= 1) sum += __shfl_xor(sum, s, 32);
  const float inv = 1.0f / sum;
  _Float16* orow = Y + ((size_t)b * CC + i) * CC;
#pragma unroll
  for (int t = 0; t < 32; ++t) orow[lane + t * 32] = (_Float16)(v[t] * inv);
}

// ---------------------------------------------------------------------------
// K3: column softmax, written TRANSPOSED so it is N-major for the GEMM:
//   P1t[b][j][k] = softmax_k( mask[b,k] ? B_t[b,k,j] : NEG )
// Mask reads are wave-uniform -> scalar loads. grid (C/256, BATCH), block 256.
// ---------------------------------------------------------------------------
__global__ void k_col_softmax(const float* __restrict__ Bt, const int* __restrict__ mask,
                              _Float16* __restrict__ P1t) {
  const int j = blockIdx.x * 256 + threadIdx.x;
  const int b = blockIdx.y;
  const float* col  = Bt + (size_t)b * CC * CC + j;
  const int*   mrow = mask + (size_t)b * CC;
  float m = NEGF;
  for (int k = 0; k < CC; ++k) {
    float v = (mrow[k] != 0) ? col[(size_t)k * CC] : NEGF;
    m = fmaxf(m, v);
  }
  float sum = 0.0f;
  for (int k = 0; k < CC; ++k) {
    float v = (mrow[k] != 0) ? col[(size_t)k * CC] : NEGF;
    sum += __expf(v - m);
  }
  const float inv = 1.0f / sum;
  _Float16* out = P1t + ((size_t)b * CC + j) * CC;
  for (int k = 0; k < CC; k += 8) {
    v8h pack;
#pragma unroll
    for (int t = 0; t < 8; ++t) {
      float v = (mrow[k + t] != 0) ? col[(size_t)(k + t) * CC] : NEGF;
      pack[t] = (_Float16)(__expf(v - m) * inv);
    }
    *reinterpret_cast<v8h*>(out + k) = pack;
  }
}

// ---------------------------------------------------------------------------
// K4: B_new = h1 @ h2^T + gamma * (B_t_2 @ B_t_1), diagonal zeroed. f32 out.
// Dominant GEMM. Each wave computes a 2x2 grid of 16x16 tiles (32x32 output):
// per K-step, 4 fragment loads feed 4 WMMAs (1 KB/WMMA of L2 traffic).
// block 256 = 8 waves as 2(M)x4(N); macro tile 64x128. grid (8, 16, 16).
// Working set (P2+P1t = 64 MB f16) lives in the 192 MB L2.
// ---------------------------------------------------------------------------
__global__ void k_bnew(const _Float16* __restrict__ h1h, const _Float16* __restrict__ h2h,
                       const _Float16* __restrict__ P2,  const _Float16* __restrict__ P1t,
                       const float* __restrict__ gamma_p, float* __restrict__ Bnew) {
  const int wave = threadIdx.x >> 5;
  const int wm = wave >> 2, wn = wave & 3;
  const int b  = blockIdx.z;
  const int i0 = blockIdx.y * 64 + wm * 32;    // wave rows [i0, i0+32)
  const int j0 = blockIdx.x * 128 + wn * 32;   // wave cols [j0, j0+32)
  const size_t base = (size_t)b * CC * CC;

  // gamma * (B_t_2 @ B_t_1): K = 1024
  const _Float16* Ab = P2  + base + (size_t)i0 * CC;   // A: (i, k) row-major
  const _Float16* Bb = P1t + base + (size_t)j0 * CC;   // B: N-major (j, k)
  v8f c00 = {}, c01 = {}, c10 = {}, c11 = {};
#pragma unroll 2
  for (int k = 0; k < CC; k += 32) {
    const v16h a0 = load_frag16(Ab + k, CC);
    const v16h a1 = load_frag16(Ab + (size_t)16 * CC + k, CC);
    const v16h b0 = load_frag16(Bb + k, CC);
    const v16h b1 = load_frag16(Bb + (size_t)16 * CC + k, CC);
    c00 = wmma(a0, b0, c00);  c01 = wmma(a0, b1, c01);
    c10 = wmma(a1, b0, c10);  c11 = wmma(a1, b1, c11);
  }

  // h1 @ h2^T: K = 128
  const _Float16* A0 = h1h + (size_t)b * CC * DD + (size_t)i0 * DD;
  const _Float16* B0 = h2h + (size_t)b * CC * DD + (size_t)j0 * DD;
  v8f d00 = {}, d01 = {}, d10 = {}, d11 = {};
#pragma unroll
  for (int k = 0; k < DD; k += 32) {
    const v16h a0 = load_frag16(A0 + k, DD);
    const v16h a1 = load_frag16(A0 + 16 * DD + k, DD);
    const v16h b0 = load_frag16(B0 + k, DD);
    const v16h b1 = load_frag16(B0 + 16 * DD + k, DD);
    d00 = wmma(a0, b0, d00);  d01 = wmma(a0, b1, d01);
    d10 = wmma(a1, b0, d10);  d11 = wmma(a1, b1, d11);
  }

  const float g = *gamma_p;
  TILE_EPILOGUE(lane, n, mh)
  float* out = Bnew + base + (size_t)i0 * CC + j0;
#pragma unroll
  for (int v = 0; v < 8; ++v) {
    const int rA = mh + v, rB = 16 + mh + v;
    float val;
    val = d00[v] + g * c00[v];
    out[(size_t)rA * CC + n]      = (i0 + rA == j0 + n)      ? 0.0f : val;
    val = d01[v] + g * c01[v];
    out[(size_t)rA * CC + 16 + n] = (i0 + rA == j0 + 16 + n) ? 0.0f : val;
    val = d10[v] + g * c10[v];
    out[(size_t)rB * CC + n]      = (i0 + rB == j0 + n)      ? 0.0f : val;
    val = d11[v] + g * c11[v];
    out[(size_t)rB * CC + 16 + n] = (i0 + rB == j0 + 16 + n) ? 0.0f : val;
  }
}

// ---------------------------------------------------------------------------
// K5: h_tmp = B_tt @ h_b, written f16 as (c,b,d) for the SFU A-operand.
// 2x2 wave tiling: wave = 32(i) x 32(d). block 256 = 8 waves (2 M x 4 N) ->
// 64(i) x 128(d) macro tile. grid (C/64, BATCH).
// ---------------------------------------------------------------------------
__global__ void k_htmp(const _Float16* __restrict__ Btt,  // (b,i,j) A
                       const _Float16* __restrict__ hT,   // (b,d,j) N-major B
                       _Float16* __restrict__ htmp) {     // (c,b,d)
  const int wave = threadIdx.x >> 5;
  const int wm = wave >> 2, wn = wave & 3;
  const int b  = blockIdx.y;
  const int i0 = blockIdx.x * 64 + wm * 32;
  const int d0 = wn * 32;
  const _Float16* Ab = Btt + ((size_t)b * CC + i0) * CC;
  const _Float16* Bb = hT  + ((size_t)b * DD + d0) * CC;
  v8f c00 = {}, c01 = {}, c10 = {}, c11 = {};
#pragma unroll 2
  for (int k = 0; k < CC; k += 32) {
    const v16h a0 = load_frag16(Ab + k, CC);
    const v16h a1 = load_frag16(Ab + (size_t)16 * CC + k, CC);
    const v16h b0 = load_frag16(Bb + k, CC);
    const v16h b1 = load_frag16(Bb + (size_t)16 * CC + k, CC);
    c00 = wmma(a0, b0, c00);  c01 = wmma(a0, b1, c01);
    c10 = wmma(a1, b0, c10);  c11 = wmma(a1, b1, c11);
  }
  TILE_EPILOGUE(lane, n, mh)
#pragma unroll
  for (int v = 0; v < 8; ++v) {
    const int iA = i0 + mh + v, iB = i0 + 16 + mh + v;
    htmp[((size_t)iA * BATCH + b) * DD + d0 + n]      = (_Float16)c00[v];
    htmp[((size_t)iA * BATCH + b) * DD + d0 + 16 + n] = (_Float16)c01[v];
    htmp[((size_t)iB * BATCH + b) * DD + d0 + n]      = (_Float16)c10[v];
    htmp[((size_t)iB * BATCH + b) * DD + d0 + 16 + n] = (_Float16)c11[v];
  }
}

// ---------------------------------------------------------------------------
// K6: SFU gate. m = [h | h_tmp | h*h_tmp | h-h_tmp] built on the fly inside
// the A-fragment loader (each 8-elem K-chunk lies in one 128-wide segment).
//   x = relu(m @ Wr^T), g = sigmoid(m @ Wg^T), o = x*g + (1-g)*h
// Two A-row tiles share the Wr/Wg B-fragments (4 WMMA per K-step).
// block 256 = 8 waves over d-tiles (N=128), M = 32 (c,b)-rows. grid 512.
// ---------------------------------------------------------------------------
__device__ __forceinline__ v8h m_chunk(const _Float16* __restrict__ hrow,
                                       const _Float16* __restrict__ trow, int q) {
  const int seg = q >> 7, off = q & (DD - 1);
  const v8h hv = *reinterpret_cast<const v8h*>(hrow + off);
  const v8h tv = *reinterpret_cast<const v8h*>(trow + off);
  v8h r;
  if (seg == 0)      r = hv;
  else if (seg == 1) r = tv;
  else if (seg == 2) {
#pragma unroll
    for (int i = 0; i < 8; ++i) r[i] = hv[i] * tv[i];
  } else {
#pragma unroll
    for (int i = 0; i < 8; ++i) r[i] = hv[i] - tv[i];
  }
  return r;
}

__device__ __forceinline__ v16h make_m_frag(const _Float16* __restrict__ hrow,
                                            const _Float16* __restrict__ trow,
                                            int k, int kb) {
  const v8h lo = m_chunk(hrow, trow, k + kb);
  const v8h hi = m_chunk(hrow, trow, k + kb + 16);
  v16h a;
#pragma unroll
  for (int i = 0; i < 8; ++i) { a[i] = lo[i]; a[8 + i] = hi[i]; }
  return a;
}

__global__ void k_sfu(const _Float16* __restrict__ hh,    // (r=(c,b), d) f16
                      const _Float16* __restrict__ htmp,  // (r, d) f16
                      const _Float16* __restrict__ Wrh,   // (d_out, k) N-major, ld 512
                      const _Float16* __restrict__ Wgh,
                      const float* __restrict__ hf,       // original h f32
                      float* __restrict__ o) {
  const int wave = threadIdx.x >> 5;
  const int r0 = blockIdx.x * 32;
  const int n0 = wave * 16;
  const int lane = threadIdx.x & 31;
  const int arow = lane & 15;          // == epilogue n
  const int kb   = (lane >> 4) << 3;   // == epilogue mh
  const _Float16* hrowA = hh   + (size_t)(r0 + arow) * DD;
  const _Float16* trowA = htmp + (size_t)(r0 + arow) * DD;
  const _Float16* hrowB = hrowA + (size_t)16 * DD;
  const _Float16* trowB = trowA + (size_t)16 * DD;
  v8f xA = {}, gA = {}, xB = {}, gB = {};
#pragma unroll 2
  for (int k = 0; k < 4 * DD; k += 32) {
    const v16h a0 = make_m_frag(hrowA, trowA, k, kb);
    const v16h a1 = make_m_frag(hrowB, trowB, k, kb);
    const v16h wr = load_frag16(Wrh + (size_t)n0 * 4 * DD + k, 4 * DD);
    const v16h wg = load_frag16(Wgh + (size_t)n0 * 4 * DD + k, 4 * DD);
    xA = wmma(a0, wr, xA);  gA = wmma(a0, wg, gA);
    xB = wmma(a1, wr, xB);  gB = wmma(a1, wg, gB);
  }
  const int n  = arow;                 // D-tile epilogue indices (reuse lane calc)
  const int mh = kb;
#pragma unroll
  for (int v = 0; v < 8; ++v) {
    const int rA = r0 + mh + v, rB = rA + 16;
    {
      const float hval = hf[(size_t)rA * DD + n0 + n];
      const float x  = fmaxf(xA[v], 0.0f);
      const float gg = 1.0f / (1.0f + __expf(-gA[v]));
      o[(size_t)rA * DD + n0 + n] = x * gg + (1.0f - gg) * hval;
    }
    {
      const float hval = hf[(size_t)rB * DD + n0 + n];
      const float x  = fmaxf(xB[v], 0.0f);
      const float gg = 1.0f / (1.0f + __expf(-gB[v]));
      o[(size_t)rB * DD + n0 + n] = x * gg + (1.0f - gg) * hval;
    }
  }
}

// ---------------------------------------------------------------------------
// Host launcher. Inputs (dict order):
// 0 h(C,B,D) 1 h_mask(B,C) 2 B_t(B,C,C) 3 W1_w 4 W1_b 5 W2_w 6 W2_b
// 7 gamma 8 Wr_w(D,4D) 9 Wg_w(D,4D).  d_out = [o (C,B,D) | B_new (B,C,C)] f32.
// ---------------------------------------------------------------------------
extern "C" void kernel_launch(void* const* d_in, const int* in_sizes, int n_in,
                              void* d_out, int out_size, void* d_ws, size_t ws_size,
                              hipStream_t stream) {
  const float* h     = (const float*)d_in[0];
  const int*   hmask = (const int*)d_in[1];
  const float* Bt    = (const float*)d_in[2];
  const float* W1w   = (const float*)d_in[3];
  const float* W1b   = (const float*)d_in[4];
  const float* W2w   = (const float*)d_in[5];
  const float* W2b   = (const float*)d_in[6];
  const float* gam   = (const float*)d_in[7];
  const float* Wrw   = (const float*)d_in[8];
  const float* Wgw   = (const float*)d_in[9];

  float* o    = (float*)d_out;
  float* Bnew = o + (size_t)CC * BATCH * DD;

  char* ws = (char*)d_ws;
  const size_t MB = 1u << 20;
  _Float16* hh   = (_Float16*)(ws + 0 * MB);    // (c,b,d)           4 MB
  _Float16* hT   = (_Float16*)(ws + 4 * MB);    // (b,d,c)           4 MB
  _Float16* W1h  = (_Float16*)(ws + 8 * MB);    // (e,d)            32 KB
  _Float16* W2h  = (_Float16*)(ws + 8 * MB + (64u << 10));
  _Float16* Wrh  = (_Float16*)(ws + 8 * MB + (128u << 10)); // (d,4d) 128 KB
  _Float16* Wgh  = (_Float16*)(ws + 8 * MB + (256u << 10));
  _Float16* h1h  = (_Float16*)(ws + 9 * MB);    // (b,c,e)           4 MB
  _Float16* h2h  = (_Float16*)(ws + 13 * MB);   // (b,c,e)           4 MB
  _Float16* P2   = (_Float16*)(ws + 17 * MB);   // (b,i,k)          32 MB
  _Float16* P1t  = (_Float16*)(ws + 49 * MB);   // (b,j,k)          32 MB
  _Float16* Btt  = (_Float16*)(ws + 81 * MB);   // (b,i,j)          32 MB
  _Float16* htmp = (_Float16*)(ws + 113 * MB);  // (c,b,d)           4 MB

  const int NH = CC * BATCH * DD;  // 2M

  // precision staging
  k_cvt<<<dim3((NH + 255) / 256), 256, 0, stream>>>(h, hh, NH);
  k_h_transpose<<<dim3(NH / 256), 256, 0, stream>>>(h, hT);
  k_cvt<<<dim3((DD * DD + 255) / 256), 256, 0, stream>>>(W1w, W1h, DD * DD);
  k_cvt<<<dim3((DD * DD + 255) / 256), 256, 0, stream>>>(W2w, W2h, DD * DD);
  k_cvt<<<dim3((DD * 4 * DD + 255) / 256), 256, 0, stream>>>(Wrw, Wrh, DD * 4 * DD);
  k_cvt<<<dim3((DD * 4 * DD + 255) / 256), 256, 0, stream>>>(Wgw, Wgh, DD * 4 * DD);

  // h1 / h2 projections (WMMA)
  k_h12<<<dim3(CC / 16, BATCH), 256, 0, stream>>>(hh, W1h, W2h, W1b, W2b, h1h, h2h);

  // masked softmaxes of B_t (axis=2 row-wise; axis=1 column-wise, transposed out)
  k_row_softmax<<<dim3(CC / 8, BATCH), 256, 0, stream>>>(Bt, hmask, P2);
  k_col_softmax<<<dim3(CC / 256, BATCH), 256, 0, stream>>>(Bt, hmask, P1t);

  // B_new = h1 h2^T + gamma * P2 P1 (diag zeroed) -> second half of d_out
  k_bnew<<<dim3(CC / 128, CC / 64, BATCH), 256, 0, stream>>>(h1h, h2h, P2, P1t, gam, Bnew);

  // attention softmax + context mix
  k_row_softmax<<<dim3(CC / 8, BATCH), 256, 0, stream>>>(Bnew, hmask, Btt);
  k_htmp<<<dim3(CC / 64, BATCH), 256, 0, stream>>>(Btt, hT, htmp);

  // SFU gate -> first half of d_out
  k_sfu<<<dim3(CC * BATCH / 32), 256, 0, stream>>>(hh, htmp, Wrh, Wgh, h, o);
}
